// MPNN_e_4612794876383
// MI455X (gfx1250) — matrix operand order repaired and checked by reading, compiled, not run on hardware
//
#include <hip/hip_runtime.h>
#include <hip/hip_bf16.h>

typedef __attribute__((ext_vector_type(16))) _Float16 v16h;
typedef __attribute__((ext_vector_type(8)))  _Float16 v8h;
typedef __attribute__((ext_vector_type(8)))  float    v8f;

#define C 64
#define KSTEPS_EDGE 130      // (64*64 + 64) / 32  : nn_w plus nn_b rows
#define XJ_STRIDE   72       // halves; 144B rows -> 16B aligned, conflict-spread
#define EA_STRIDE   68       // halves; (m*68)*2/4 = m*34 mod 64 distinct
#define WPB         8        // waves per block (node kernel, 256 threads)
#define EWPB        4        // waves per block (edge kernel, 128 threads)

__device__ __forceinline__ v8f wmma_f16(v16h a, v16h b, v8f c) {
  // (neg_a, A, neg_b, B, c_mod, C, reuse_a, reuse_b)
  return __builtin_amdgcn_wmma_f32_16x16x32_f16(false, a, false, b, (short)0, c,
                                                false, false);
}

// ---------------------------------------------------------------------------
// Swizzle nn_w [64, 4096] (+ nn_b [4096]) into WMMA B-fragments, f16.
// Extended K = 4160: rows r<4096 -> nn_w[r>>6][ (r&63)*64 + o ], rows >=4096 -> nn_b.
// Fragment f = t*4+nt; lane l element e: K = 32t + (e<8 ? e : 8+e) + (l<16 ? 0 : 8),
// N = nt*16 + (l&15). Stored at f*512 + l*16 + e.
// ---------------------------------------------------------------------------
__global__ void swizzle_nn_kernel(const float* __restrict__ nn_w,
                                  const float* __restrict__ nn_b,
                                  _Float16* __restrict__ Bswz, int total) {
  int tid = blockIdx.x * blockDim.x + threadIdx.x;
  if (tid >= total) return;
  int f = tid >> 9;
  int w = tid & 511;
  int l = w >> 4;
  int e = w & 15;
  int t  = f >> 2;
  int nt = f & 3;
  int Kg = 32 * t + ((e < 8) ? e : 8 + e) + ((l < 16) ? 0 : 8);
  int Ng = nt * 16 + (l & 15);
  int k = Kg >> 6;
  int i = Kg & 63;
  float v = (k < C) ? nn_w[(size_t)k * (C * C) + i * C + Ng]
                    : nn_b[(size_t)i * C + Ng];
  Bswz[tid] = (_Float16)v;
}

// Swizzle a plain [64, 64] row-major matrix (lin_w / fin_w): K = 64 -> 8 frags.
__global__ void swizzle_sq_kernel(const float* __restrict__ Wsrc,
                                  _Float16* __restrict__ out) {
  int tid = blockIdx.x * blockDim.x + threadIdx.x;
  if (tid >= 8 * 512) return;
  int f = tid >> 9;
  int w = tid & 511;
  int l = w >> 4;
  int e = w & 15;
  int t  = f >> 2;
  int nt = f & 3;
  int Kg = 32 * t + ((e < 8) ? e : 8 + e) + ((l < 16) ? 0 : 8);
  int Ng = nt * 16 + (l & 15);
  out[tid] = (_Float16)Wsrc[(size_t)Kg * C + Ng];
}

__global__ void zero_kernel(float* __restrict__ p, int n) {
  int i = blockIdx.x * blockDim.x + threadIdx.x;
  if (i < n) p[i] = 0.0f;
}

// ---------------------------------------------------------------------------
// Edge message kernel: per-wave 32-edge tile (2 WMMA M-tiles sharing the B
// stream), K = 4160 fused GEMM msg = (ea outer x_j) @ Bswz, then f32-atomic
// scatter-add into acc[dst].
// ---------------------------------------------------------------------------
__global__ __launch_bounds__(128, 1)
void edge_msg_kernel(const float* __restrict__ x,        // [N, 64] f32
                     const float* __restrict__ ea,       // [E, 64] f32
                     const int*   __restrict__ src,
                     const int*   __restrict__ dst,
                     const _Float16* __restrict__ Bswz,  // 130*4 fragments
                     float* __restrict__ acc,            // [N, 64] f32 (zeroed)
                     int E_, int N_) {
  __shared__ _Float16 xj_lds[EWPB][32 * XJ_STRIDE];
  __shared__ _Float16 ea_lds[EWPB][32 * EA_STRIDE];
  __shared__ int      src_lds[EWPB][32];
  __shared__ int      dst_lds[EWPB][32];

  const int lane = threadIdx.x & 31;
  const int wv   = threadIdx.x >> 5;
  const int ntile = (E_ + 31) >> 5;
  const int tile  = blockIdx.x * EWPB + wv;
  const bool valid = tile < ntile;
  const int e0 = tile << 5;

  // Phase 1: edge indices (one per lane)
  if (valid) {
    int e = e0 + lane;
    src_lds[wv][lane] = (e < E_) ? src[e] : 0;
    dst_lds[wv][lane] = (e < E_) ? dst[e] : 0;
  }
  __syncthreads();

  // Phase 2: stage gathered x_j and edge_attr, both f16
  if (valid) {
    for (int idx = lane; idx < 32 * C; idx += 32) {
      int row = idx >> 6;
      int col = idx & 63;
      int e = e0 + row;
      float xv = 0.0f, ev = 0.0f;
      if (e < E_) {
        ev = ea[(size_t)e * C + col];
        xv = x[(size_t)src_lds[wv][row] * C + col];
      }
      xj_lds[wv][row * XJ_STRIDE + col] = (_Float16)xv;
      ea_lds[wv][row * EA_STRIDE + col] = (_Float16)ev;
    }
    ea_lds[wv][lane * EA_STRIDE + 64] = (_Float16)1.0f;  // nn_b virtual column
  }
  __syncthreads();

  if (!valid) return;

  const int m  = lane & 15;                 // A-row within M-tile
  const int hs = (lane < 16) ? 0 : 8;       // K-half select per A layout
  const _Float16* xr0 = &xj_lds[wv][m * XJ_STRIDE];
  const _Float16* xr1 = &xj_lds[wv][(m + 16) * XJ_STRIDE];
  const _Float16* er0 = &ea_lds[wv][m * EA_STRIDE];
  const _Float16* er1 = &ea_lds[wv][(m + 16) * EA_STRIDE];
  const v16h* Bf = (const v16h*)Bswz;

  v8f accf[2][4] = {{v8f{}, v8f{}, v8f{}, v8f{}},
                    {v8f{}, v8f{}, v8f{}, v8f{}}};

  for (int t = 0; t < KSTEPS_EDGE; ++t) {
    int kblk  = t >> 1;                     // edge_attr column (0..64)
    int ibase = (t & 1) << 5;               // x_j column window base (0 or 32)
    int i0 = ibase + hs;
    _Float16 s0 = er0[kblk];
    _Float16 s1 = er1[kblk];
    v8h a00 = *(const v8h*)(xr0 + i0)      * s0;
    v8h a01 = *(const v8h*)(xr0 + i0 + 16) * s0;
    v8h a10 = *(const v8h*)(xr1 + i0)      * s1;
    v8h a11 = *(const v8h*)(xr1 + i0 + 16) * s1;
    v16h A0 = __builtin_shufflevector(a00, a01, 0, 1, 2, 3, 4, 5, 6, 7,
                                                8, 9, 10, 11, 12, 13, 14, 15);
    v16h A1 = __builtin_shufflevector(a10, a11, 0, 1, 2, 3, 4, 5, 6, 7,
                                                8, 9, 10, 11, 12, 13, 14, 15);
    int fb = t * 128 + lane;                // 32 v16h per fragment
    v16h b0 = Bf[fb];
    v16h b1 = Bf[fb + 32];
    v16h b2 = Bf[fb + 64];
    v16h b3 = Bf[fb + 96];
    accf[0][0] = wmma_f16(A0, b0, accf[0][0]);
    accf[1][0] = wmma_f16(A1, b0, accf[1][0]);
    accf[0][1] = wmma_f16(A0, b1, accf[0][1]);
    accf[1][1] = wmma_f16(A1, b1, accf[1][1]);
    accf[0][2] = wmma_f16(A0, b2, accf[0][2]);
    accf[1][2] = wmma_f16(A1, b2, accf[1][2]);
    accf[0][3] = wmma_f16(A0, b3, accf[0][3]);
    accf[1][3] = wmma_f16(A1, b3, accf[1][3]);
  }

  // Scatter: C/D layout -> lane<16: N=lane, M=r; lane>=16: N=lane-16, M=r+8
  const int myN = lane & 15;
  const int mb  = (lane < 16) ? 0 : 8;
#pragma unroll
  for (int g = 0; g < 2; ++g) {
#pragma unroll
    for (int nt = 0; nt < 4; ++nt) {
      v8f cc = accf[g][nt];
#pragma unroll
      for (int r = 0; r < 8; ++r) {
        int row = g * 16 + mb + r;
        int e = e0 + row;
        if (e < E_) {
          int d = dst_lds[wv][row];
          atomicAdd(&acc[(size_t)d * C + nt * 16 + myN], cc[r]);
        }
      }
    }
  }
}

// ---------------------------------------------------------------------------
// Node update: out = [relu]( msg_acc + x @ Wswz + bias ). WMMA K = 64.
// ---------------------------------------------------------------------------
__global__ __launch_bounds__(256, 1)
void node_update_kernel(const float* __restrict__ x,        // [N, 64]
                        const _Float16* __restrict__ Wswz,  // 8 fragments
                        const float* __restrict__ bias,     // [64]
                        const float* __restrict__ msg,      // [N, 64] or null
                        float* __restrict__ out,            // [N, 64]
                        int N_, int do_relu) {
  __shared__ _Float16 xl[WPB][16 * XJ_STRIDE];

  const int lane = threadIdx.x & 31;
  const int wv   = threadIdx.x >> 5;
  const int ntile = (N_ + 15) >> 4;
  const int tile  = blockIdx.x * WPB + wv;
  const bool valid = tile < ntile;
  const int n0 = tile << 4;

  if (valid) {
    for (int idx = lane; idx < 16 * C; idx += 32) {
      int row = idx >> 6;
      int col = idx & 63;
      int n = n0 + row;
      float xv = (n < N_) ? x[(size_t)n * C + col] : 0.0f;
      xl[wv][row * XJ_STRIDE + col] = (_Float16)xv;
    }
  }
  __syncthreads();
  if (!valid) return;

  const int m  = lane & 15;
  const int hs = (lane < 16) ? 0 : 8;
  const _Float16* xrow = &xl[wv][m * XJ_STRIDE];
  const v16h* Bf = (const v16h*)Wswz;

  v8f accf[4] = {v8f{}, v8f{}, v8f{}, v8f{}};
#pragma unroll
  for (int t = 0; t < 2; ++t) {
    int i0 = (t << 5) + hs;
    v8h c0 = *(const v8h*)(xrow + i0);
    v8h c1 = *(const v8h*)(xrow + i0 + 16);
    v16h a = __builtin_shufflevector(c0, c1, 0, 1, 2, 3, 4, 5, 6, 7,
                                             8, 9, 10, 11, 12, 13, 14, 15);
    int fb = t * 128 + lane;
    accf[0] = wmma_f16(a, Bf[fb],      accf[0]);
    accf[1] = wmma_f16(a, Bf[fb + 32], accf[1]);
    accf[2] = wmma_f16(a, Bf[fb + 64], accf[2]);
    accf[3] = wmma_f16(a, Bf[fb + 96], accf[3]);
  }

  const int myN = lane & 15;
  const int mb  = (lane < 16) ? 0 : 8;
#pragma unroll
  for (int nt = 0; nt < 4; ++nt) {
    v8f cc = accf[nt];
#pragma unroll
    for (int r = 0; r < 8; ++r) {
      int n = n0 + mb + r;
      if (n < N_) {
        int col = nt * 16 + myN;
        float v = cc[r];
        if (msg)  v += msg[(size_t)n * C + col];
        v += bias[col];
        if (do_relu) v = fmaxf(v, 0.0f);
        out[(size_t)n * C + col] = v;
      }
    }
  }
}

// ---------------------------------------------------------------------------
extern "C" void kernel_launch(void* const* d_in, const int* in_sizes, int n_in,
                              void* d_out, int out_size, void* d_ws, size_t ws_size,
                              hipStream_t stream) {
  const float* feature    = (const float*)d_in[0];
  const int*   edge_index = (const int*)  d_in[1];
  const float* edge_attr  = (const float*)d_in[2];
  const float* nn_w0  = (const float*)d_in[3];
  const float* nn_b0  = (const float*)d_in[4];
  const float* lin_w0 = (const float*)d_in[5];
  const float* bias0  = (const float*)d_in[6];
  const float* nn_w1  = (const float*)d_in[7];
  const float* nn_b1  = (const float*)d_in[8];
  const float* lin_w1 = (const float*)d_in[9];
  const float* bias1  = (const float*)d_in[10];
  const float* fin_w  = (const float*)d_in[11];
  const float* fin_b  = (const float*)d_in[12];

  const int N_ = in_sizes[0] / C;
  const int E_ = in_sizes[1] / 2;
  const int* src = edge_index;
  const int* dst = edge_index + E_;

  // workspace carve-up (256B aligned chunks)
  char* ws = (char*)d_ws;
  auto carve = [&](size_t bytes) -> char* {
    char* p = ws;
    ws += (bytes + 255) & ~(size_t)255;
    return p;
  };
  float*    acc = (float*)   carve((size_t)N_ * C * sizeof(float));
  float*    x1  = (float*)   carve((size_t)N_ * C * sizeof(float));
  float*    x2  = (float*)   carve((size_t)N_ * C * sizeof(float));
  _Float16* B0  = (_Float16*)carve((size_t)KSTEPS_EDGE * 4 * 512 * 2);
  _Float16* B1  = (_Float16*)carve((size_t)KSTEPS_EDGE * 4 * 512 * 2);
  _Float16* L0  = (_Float16*)carve(8 * 512 * 2);
  _Float16* L1  = (_Float16*)carve(8 * 512 * 2);
  _Float16* LF  = (_Float16*)carve(8 * 512 * 2);

  const int BT = 256;
  const int totB = KSTEPS_EDGE * 4 * 512;
  swizzle_nn_kernel<<<(totB + BT - 1) / BT, BT, 0, stream>>>(nn_w0, nn_b0, B0, totB);
  swizzle_nn_kernel<<<(totB + BT - 1) / BT, BT, 0, stream>>>(nn_w1, nn_b1, B1, totB);
  swizzle_sq_kernel<<<(8 * 512) / BT, BT, 0, stream>>>(lin_w0, L0);
  swizzle_sq_kernel<<<(8 * 512) / BT, BT, 0, stream>>>(lin_w1, L1);
  swizzle_sq_kernel<<<(8 * 512) / BT, BT, 0, stream>>>(fin_w, LF);

  const int etiles  = (E_ + 31) / 32;
  const int eblocks = (etiles + EWPB - 1) / EWPB;
  const int ntiles  = (N_ + 15) / 16;
  const int nblocks = (ntiles + WPB - 1) / WPB;
  const int nzero   = (N_ * C + BT - 1) / BT;

  // layer 0
  zero_kernel<<<nzero, BT, 0, stream>>>(acc, N_ * C);
  edge_msg_kernel<<<eblocks, 128, 0, stream>>>(feature, edge_attr, src, dst, B0, acc, E_, N_);
  node_update_kernel<<<nblocks, 256, 0, stream>>>(feature, L0, bias0, acc, x1, N_, 1);

  // layer 1
  zero_kernel<<<nzero, BT, 0, stream>>>(acc, N_ * C);
  edge_msg_kernel<<<eblocks, 128, 0, stream>>>(x1, edge_attr, src, dst, B1, acc, E_, N_);
  node_update_kernel<<<nblocks, 256, 0, stream>>>(x1, L1, bias1, acc, x2, N_, 1);

  // final linear (no relu, no message accumulator)
  node_update_kernel<<<nblocks, 256, 0, stream>>>(x2, LF, fin_b, nullptr, (float*)d_out, N_, 0);
}
